// FEM0225_55439437857353
// MI455X (gfx1250) — compile-verified, matrix operand order checked by reading, not run on hardware
//
#include <hip/hip_runtime.h>
#include <hip/hip_bf16.h>

// ---------------------------------------------------------------------------
// CDNA5 (gfx1250) implementation: bf16 WMMA for every large GEMM,
// flash-attention online softmax (A_s never materialized), and
// async-to-LDS staging (ASYNCcnt path) for the attention K/V tiles.
// ---------------------------------------------------------------------------

typedef __attribute__((ext_vector_type(16))) __bf16 v16bf;
typedef __attribute__((ext_vector_type(8)))  float  v8f;
typedef __attribute__((ext_vector_type(4)))  unsigned int uint4v;

union Frag {
    v16bf  bf;
    uint4v q[2];
};

#define NB 4
#define CIN 256
#define CI 128
#define NSP 4096   // 64*64
#define EPSV 1e-5f

__device__ __forceinline__ unsigned short f2bf(float f) {
    unsigned int u = __builtin_bit_cast(unsigned int, f);
    unsigned int r = u + 0x7FFFu + ((u >> 16) & 1u);   // round to nearest even
    return (unsigned short)(r >> 16);
}
__device__ __forceinline__ float bf2f(unsigned short u) {
    return __builtin_bit_cast(float, ((unsigned int)u) << 16);
}

__device__ __forceinline__ v8f wmma_bf16(const Frag& a, const Frag& b, v8f c) {
    return __builtin_amdgcn_wmma_f32_16x16x32_bf16(
        false, a.bf, false, b.bf, (short)0, c, false, false);
}

// Generic (flat) shared-memory pointer -> raw LDS byte offset.
// ISA 10.2: in the LDS aperture the LDS address is the low 32 bits of the flat address.
__device__ __forceinline__ unsigned lds_off(const void* p) {
    return (unsigned)(size_t)p;
}

// GLOBAL_LOAD_ASYNC_TO_LDS_B128, GV mode: per-lane 16B memory -> LDS.
// Tracked with ASYNCcnt (ISA 10 / 15.18.3, opcode 98).
__device__ __forceinline__ void async_b128(unsigned lds_addr, const void* g) {
    asm volatile("global_load_async_to_lds_b128 %0, %1, off"
                 :: "v"(lds_addr),
                    "v"((unsigned long long)(size_t)g)
                 : "memory");
}
__device__ __forceinline__ void wait_async0() {
    asm volatile("s_wait_asynccnt 0" ::: "memory");
}

// ===========================================================================
// Kernel 1: all 1x1 convs as tiled WMMA GEMMs.
// Y[o,n] = sum_c W[o,c] X[c,n] + bias   (optionally BN)
// Grid: (ntile=32, app=6, batch=4), block 256 (8 waves).
// apps 0..3 write bf16 [n,c] transposed operands for attention,
// apps 4..5 write fp32 [c,n] BN-transformed residuals.
// ===========================================================================
__global__ __launch_bounds__(256) void k_conv1x1(
    const float* __restrict__ qin, const float* __restrict__ sin,
    const float* w_v, const float* b_v,
    const float* w_k, const float* b_k,
    const float* w_q, const float* b_q,
    const float* w_ts, const float* b_ts,
    const float* g_ts, const float* be_ts, const float* m_ts, const float* v_ts,
    const float* w_tq, const float* b_tq,
    const float* g_tq, const float* be_tq, const float* m_tq, const float* v_tq,
    unsigned short* vqT, unsigned short* vsT,
    unsigned short* kxT, unsigned short* qxT,
    float* s_t, float* q_t)
{
    __shared__ unsigned short Wb[128 * 32];
    __shared__ unsigned short Xb[128 * 32];

    const int ntile = blockIdx.x;
    const int app   = blockIdx.y;
    const int b     = blockIdx.z;
    const int tid   = threadIdx.x;
    const int wave  = tid >> 5;
    const int lane  = tid & 31;
    const int hoff  = (lane >= 16) ? 8 : 0;

    const float *X = nullptr, *W = nullptr, *Bias = nullptr;
    const float *bng = nullptr, *bnb = nullptr, *bnm = nullptr, *bnv = nullptr;
    unsigned short* obf = nullptr;
    float* of32 = nullptr;
    switch (app) {
        case 0: X = qin; W = w_v;  Bias = b_v;  obf = vqT; break;
        case 1: X = sin; W = w_v;  Bias = b_v;  obf = vsT; break;
        case 2: X = sin; W = w_k;  Bias = b_k;  obf = kxT; break;
        case 3: X = qin; W = w_q;  Bias = b_q;  obf = qxT; break;
        case 4: X = sin; W = w_ts; Bias = b_ts; of32 = s_t;
                bng = g_ts; bnb = be_ts; bnm = m_ts; bnv = v_ts; break;
        default:X = qin; W = w_tq; Bias = b_tq; of32 = q_t;
                bng = g_tq; bnb = be_tq; bnm = m_tq; bnv = v_tq; break;
    }

    v8f acc[8];
    const v8f vzero = {0.f,0.f,0.f,0.f,0.f,0.f,0.f,0.f};
    for (int i = 0; i < 8; ++i) acc[i] = vzero;

    for (int kc = 0; kc < 8; ++kc) {          // K = 256 in chunks of 32
        {   // stage W chunk [128 o][32 c] -> bf16 LDS
            const int o  = tid >> 1;
            const int c0 = (tid & 1) * 16;
            const float* src = W + (size_t)o * CIN + kc * 32 + c0;
            #pragma unroll
            for (int j = 0; j < 16; ++j) Wb[o * 32 + c0 + j] = f2bf(src[j]);
        }
        {   // stage X chunk [32 c][128 n] -> LDS transposed [n][c] bf16
            const int cc  = tid >> 3;
            const int nn0 = (tid & 7) * 16;
            const float* src = X + (size_t)b * CIN * NSP
                                 + (size_t)(kc * 32 + cc) * NSP
                                 + (size_t)ntile * 128 + nn0;
            #pragma unroll
            for (int j = 0; j < 16; ++j) Xb[(nn0 + j) * 32 + cc] = f2bf(src[j]);
        }
        __syncthreads();

        Frag a;
        {
            const int o = wave * 16 + (lane & 15);
            a.q[0] = *(const uint4v*)&Wb[o * 32 + hoff];
            a.q[1] = *(const uint4v*)&Wb[o * 32 + hoff + 16];
        }
        #pragma unroll
        for (int nt = 0; nt < 8; ++nt) {
            Frag bb;
            const int n = nt * 16 + (lane & 15);
            bb.q[0] = *(const uint4v*)&Xb[n * 32 + hoff];
            bb.q[1] = *(const uint4v*)&Xb[n * 32 + hoff + 16];
            acc[nt] = wmma_bf16(a, bb, acc[nt]);
        }
        __syncthreads();
    }

    #pragma unroll
    for (int e = 0; e < 8; ++e) {
        const int o = wave * 16 + e + hoff;
        const float bi = Bias[o];
        float sc = 1.f, sh = 0.f;
        if (of32) {
            sc = bng[o] * __frsqrt_rn(bnv[o] + EPSV);
            sh = bnb[o] - bnm[o] * sc;
        }
        #pragma unroll
        for (int nt = 0; nt < 8; ++nt) {
            const int n = ntile * 128 + nt * 16 + (lane & 15);
            float val = acc[nt][e] + bi;
            if (of32) {
                val = val * sc + sh;
                of32[((size_t)b * CI + o) * NSP + n] = val;
            } else {
                obf[((size_t)b * NSP + n) * CI + o] = f2bf(val);
            }
        }
    }
}

// ===========================================================================
// Kernel 2: SE channel gates (avg pool -> 128->8 relu -> 8->128 sigmoid)
// Grid: (gate 2, batch 4), block 128.
// ===========================================================================
__global__ __launch_bounds__(128) void k_gate(
    const unsigned short* __restrict__ vsT, const unsigned short* __restrict__ vqT,
    const float* w1a, const float* b1a, const float* w2a, const float* b2a,
    const float* w1b, const float* b1b, const float* w2b, const float* b2b,
    float* gate1, float* gate2)
{
    const int gid = blockIdx.x;
    const int b   = blockIdx.y;
    const int c   = threadIdx.x;

    const unsigned short* X = (gid == 0) ? vsT : vqT;
    const float* w1 = (gid == 0) ? w1a : w1b;
    const float* b1 = (gid == 0) ? b1a : b1b;
    const float* w2 = (gid == 0) ? w2a : w2b;
    const float* b2 = (gid == 0) ? b2a : b2b;

    __shared__ float p[128];
    __shared__ float h[8];

    float sum = 0.f;
    const unsigned short* base = X + (size_t)b * NSP * CI + c;
    for (int n = 0; n < NSP; ++n) sum += bf2f(base[(size_t)n * CI]);
    p[c] = sum * (1.f / (float)NSP);
    __syncthreads();

    if (c < 8) {
        float a = b1[c];
        for (int j = 0; j < 128; ++j) a += p[j] * w1[c * 128 + j];
        h[c] = fmaxf(a, 0.f);
    }
    __syncthreads();

    float a = b2[c];
    #pragma unroll
    for (int j = 0; j < 8; ++j) a += h[j] * w2[c * 8 + j];
    ((gid == 0) ? gate1 : gate2)[b * CI + c] = 1.f / (1.f + __expf(-a));
}

// ===========================================================================
// Kernel 3: flash attention (online softmax), two passes.
//   pass 0: Q=kxT K=qxT V=vsT -> p_s[n,c]
//   pass 1: Q=qxT K=kxT V=vqT -> q_s[n,c]
// Grid: (qtile 64, pass 2, batch 4), block 128 (4 waves, 16 queries each).
// K and V tiles staged with GLOBAL_LOAD_ASYNC_TO_LDS_B128 (ASYNCcnt).
// ===========================================================================
__global__ __launch_bounds__(128) void k_attn(
    const unsigned short* __restrict__ kxT, const unsigned short* __restrict__ qxT,
    const unsigned short* __restrict__ vsT, const unsigned short* __restrict__ vqT,
    float* __restrict__ p_s, float* __restrict__ q_s)
{
    __shared__ unsigned short Klds[64 * 128];   // [m][c]   (async target)
    __shared__ unsigned short Vrow[64 * 128];   // [m][c]   (async target)
    __shared__ unsigned short Vlds[128 * 64];   // [c][m]   (transposed)
    __shared__ unsigned short Plds[4][16 * 64]; // per-wave P re-layout

    const int qtile = blockIdx.x;
    const int pass  = blockIdx.y;
    const int b     = blockIdx.z;
    const int tid   = threadIdx.x;
    const int wave  = tid >> 5;
    const int lane  = tid & 31;
    const int hoff  = (lane >= 16) ? 8 : 0;

    const unsigned short* Qm = (pass == 0) ? kxT : qxT;
    const unsigned short* Km = (pass == 0) ? qxT : kxT;
    const unsigned short* Vm = (pass == 0) ? vsT : vqT;
    float* Om                = (pass == 0) ? p_s : q_s;

    // Q fragments stay in registers: 16 rows x 128 (4 K-slices of 32)
    Frag qf[4];
    {
        const int qrow = qtile * 64 + wave * 16 + (lane & 15);
        const unsigned short* qp = Qm + ((size_t)b * NSP + qrow) * CI;
        #pragma unroll
        for (int ks = 0; ks < 4; ++ks) {
            qf[ks].q[0] = *(const uint4v*)(qp + ks * 32 + hoff);
            qf[ks].q[1] = *(const uint4v*)(qp + ks * 32 + hoff + 16);
        }
    }

    v8f o_acc[8];
    const v8f vzero = {0.f,0.f,0.f,0.f,0.f,0.f,0.f,0.f};
    #pragma unroll
    for (int i = 0; i < 8; ++i) o_acc[i] = vzero;
    float m_i[8], l_i[8];
    #pragma unroll
    for (int r = 0; r < 8; ++r) { m_i[r] = -1e30f; l_i[r] = 0.f; }

    const unsigned kbase_lds = lds_off(&Klds[0]);
    const unsigned vbase_lds = lds_off(&Vrow[0]);

    for (int kt = 0; kt < 64; ++kt) {          // 4096 keys in chunks of 64
        // ---- async stage K and V chunks (16 KB each) into LDS -------------
        {
            const unsigned short* ksrc = Km + ((size_t)b * NSP + kt * 64) * CI;
            const unsigned short* vsrc = Vm + ((size_t)b * NSP + kt * 64) * CI;
            #pragma unroll
            for (int j = 0; j < 8; ++j) {
                const int i16 = j * 128 + tid;            // 16-byte chunk id
                async_b128(kbase_lds + i16 * 16, (const char*)ksrc + i16 * 16);
                async_b128(vbase_lds + i16 * 16, (const char*)vsrc + i16 * 16);
            }
        }
        // prefetch next chunk while async copies land
        if (kt + 1 < 64) {
            const unsigned short* nk = Km + ((size_t)b * NSP + (kt + 1) * 64) * CI;
            const unsigned short* nv = Vm + ((size_t)b * NSP + (kt + 1) * 64) * CI;
            __builtin_prefetch(nk + tid * 64, 0, 0);      // 128B/lane covers 16KB
            __builtin_prefetch(nv + tid * 64, 0, 0);
        }
        wait_async0();
        __syncthreads();        // async LDS writes visible to all waves

        // ---- LDS->LDS transpose of V: Vrow[m][c] -> Vlds[c][m] ------------
        {
            const unsigned int* vr = (const unsigned int*)Vrow;
            #pragma unroll
            for (int j = 0; j < 32; ++j) {
                const int iu = j * 128 + tid;             // uint id
                const unsigned int u = vr[iu];
                const int m = iu >> 6;                    // 64 uints per row
                const int c = (iu & 63) * 2;
                Vlds[(c + 0) * 64 + m] = (unsigned short)(u & 0xFFFFu);
                Vlds[(c + 1) * 64 + m] = (unsigned short)(u >> 16);
            }
        }
        __syncthreads();

        // ---- S = Q x K^T (4 key subtiles of 16) ---------------------------
        v8f st[4];
        #pragma unroll
        for (int mt = 0; mt < 4; ++mt) {
            v8f sa = vzero;
            #pragma unroll
            for (int ks = 0; ks < 4; ++ks) {
                Frag kf;
                const int m = mt * 16 + (lane & 15);
                kf.q[0] = *(const uint4v*)&Klds[m * 128 + ks * 32 + hoff];
                kf.q[1] = *(const uint4v*)&Klds[m * 128 + ks * 32 + hoff + 16];
                sa = wmma_bf16(qf[ks], kf, sa);
            }
            st[mt] = sa;
        }

        // ---- online softmax per query row r (q = wave*16 + r + hoff) ------
        float alpha[8];
        #pragma unroll
        for (int r = 0; r < 8; ++r) {
            float mx = st[0][r];
            #pragma unroll
            for (int mt = 1; mt < 4; ++mt) mx = fmaxf(mx, st[mt][r]);
            #pragma unroll
            for (int d = 1; d < 16; d <<= 1)
                mx = fmaxf(mx, __shfl_xor(mx, d, 32)); // stays in 16-lane half
            const float mnew = fmaxf(m_i[r], mx);
            alpha[r] = __expf(m_i[r] - mnew);
            m_i[r] = mnew;
            float rs = 0.f;
            #pragma unroll
            for (int mt = 0; mt < 4; ++mt) {
                const float pv = __expf(st[mt][r] - mnew);
                st[mt][r] = pv;
                rs += pv;
            }
            #pragma unroll
            for (int d = 1; d < 16; d <<= 1) rs += __shfl_xor(rs, d, 32);
            l_i[r] = l_i[r] * alpha[r] + rs;
        }
        #pragma unroll
        for (int ct = 0; ct < 8; ++ct)
            #pragma unroll
            for (int r = 0; r < 8; ++r) o_acc[ct][r] *= alpha[r];

        // ---- re-layout P (D-layout -> A-layout) via per-wave LDS ----------
        #pragma unroll
        for (int mt = 0; mt < 4; ++mt)
            #pragma unroll
            for (int r = 0; r < 8; ++r) {
                const int qr = r + hoff;
                const int m  = mt * 16 + (lane & 15);
                Plds[wave][qr * 64 + m] = f2bf(st[mt][r]);
            }
        asm volatile("s_wait_dscnt 0" ::: "memory"); // wave-local RAW on LDS

        Frag pa[2];
        {
            const int qr = lane & 15;
            #pragma unroll
            for (int k2 = 0; k2 < 2; ++k2) {
                pa[k2].q[0] = *(const uint4v*)&Plds[wave][qr * 64 + k2 * 32 + hoff];
                pa[k2].q[1] = *(const uint4v*)&Plds[wave][qr * 64 + k2 * 32 + hoff + 16];
            }
        }
        // ---- O += P x V ---------------------------------------------------
        #pragma unroll
        for (int ct = 0; ct < 8; ++ct) {
            #pragma unroll
            for (int k2 = 0; k2 < 2; ++k2) {
                Frag vf;
                const int c = ct * 16 + (lane & 15);
                vf.q[0] = *(const uint4v*)&Vlds[c * 64 + k2 * 32 + hoff];
                vf.q[1] = *(const uint4v*)&Vlds[c * 64 + k2 * 32 + hoff + 16];
                o_acc[ct] = wmma_bf16(pa[k2], vf, o_acc[ct]);
            }
        }
        __syncthreads();        // protect Klds/Vrow before next async stage
    }

    // ---- normalize and store [n, c] fp32 ---------------------------------
    #pragma unroll
    for (int r = 0; r < 8; ++r) {
        const float inv = 1.f / l_i[r];
        const int qg = qtile * 64 + wave * 16 + r + hoff;
        #pragma unroll
        for (int ct = 0; ct < 8; ++ct) {
            const int c = ct * 16 + (lane & 15);
            Om[((size_t)b * NSP + qg) * CI + c] = o_acc[ct][r] * inv;
        }
    }
}

// ===========================================================================
// Kernel 4: E_q = gate2*q_s + q_t ; E_s = gate1*p_s + s_t   -> d_out slots
// ===========================================================================
__global__ __launch_bounds__(256) void k_epilogue(
    const float* __restrict__ p_s, const float* __restrict__ q_s,
    const float* __restrict__ s_t, const float* __restrict__ q_t,
    const float* __restrict__ gate1, const float* __restrict__ gate2,
    float* __restrict__ eq_out, float* __restrict__ es_out)
{
    const size_t idx = (size_t)blockIdx.x * 256 + threadIdx.x; // [b][c][n]
    const int n = (int)(idx & 4095);
    const int c = (int)((idx >> 12) & 127);
    const int b = (int)(idx >> 19);
    const size_t tnc = ((size_t)b * NSP + n) * CI + c;
    if (blockIdx.y == 0) {
        eq_out[idx] = gate2[b * CI + c] * q_s[tnc] + q_t[idx];
    } else {
        es_out[idx] = gate1[b * CI + c] * p_s[tnc] + s_t[idx];
    }
}

// ===========================================================================
// Kernel 5: 3x3 conv (SAME) on cat[E_q;E_s] as 9 shifted 1x1 WMMA GEMMs,
// then BN + ReLU. Grid: (ntile 32, 1, batch 4), block 256 (8 waves).
// ===========================================================================
__global__ __launch_bounds__(256) void k_conv3x3(
    const float* __restrict__ eq, const float* __restrict__ es,
    const float* __restrict__ w_cat,
    const float* g, const float* be, const float* m, const float* v,
    float* __restrict__ out)
{
    __shared__ unsigned short Wb[128 * 32];
    __shared__ unsigned short Xb[128 * 32];

    const int ntile = blockIdx.x;
    const int b     = blockIdx.z;
    const int tid   = threadIdx.x;
    const int wave  = tid >> 5;
    const int lane  = tid & 31;
    const int hoff  = (lane >= 16) ? 8 : 0;

    v8f acc[8];
    const v8f vzero = {0.f,0.f,0.f,0.f,0.f,0.f,0.f,0.f};
    #pragma unroll
    for (int i = 0; i < 8; ++i) acc[i] = vzero;

    for (int tap = 0; tap < 9; ++tap) {
        const int dh = tap / 3 - 1, dw = tap % 3 - 1;
        for (int kc = 0; kc < 8; ++kc) {       // 256 input channels / 32
            {   // W chunk: w_cat[(o*256 + i)*9 + tap]
                const int o  = tid >> 1;
                const int c0 = (tid & 1) * 16;
                #pragma unroll
                for (int j = 0; j < 16; ++j)
                    Wb[o * 32 + c0 + j] =
                        f2bf(w_cat[((size_t)o * 256 + kc * 32 + c0 + j) * 9 + tap]);
            }
            {   // X chunk: shifted, zero-padded input tile -> [n][c] bf16
                const int cc  = tid >> 3;
                const int nn0 = (tid & 7) * 16;
                const int ich = kc * 32 + cc;
                const float* src = ((ich < 128) ? eq : es)
                                   + ((size_t)b * CI + (ich & 127)) * NSP;
                #pragma unroll
                for (int j = 0; j < 16; ++j) {
                    const int n  = ntile * 128 + nn0 + j;
                    const int hs = (n >> 6) + dh, ws = (n & 63) + dw;
                    float val = 0.f;
                    if ((unsigned)hs < 64u && (unsigned)ws < 64u)
                        val = src[(hs << 6) + ws];
                    Xb[(nn0 + j) * 32 + cc] = f2bf(val);
                }
            }
            __syncthreads();

            Frag a;
            {
                const int o = wave * 16 + (lane & 15);
                a.q[0] = *(const uint4v*)&Wb[o * 32 + hoff];
                a.q[1] = *(const uint4v*)&Wb[o * 32 + hoff + 16];
            }
            #pragma unroll
            for (int nt = 0; nt < 8; ++nt) {
                Frag bb;
                const int n = nt * 16 + (lane & 15);
                bb.q[0] = *(const uint4v*)&Xb[n * 32 + hoff];
                bb.q[1] = *(const uint4v*)&Xb[n * 32 + hoff + 16];
                acc[nt] = wmma_bf16(a, bb, acc[nt]);
            }
            __syncthreads();
        }
    }

    #pragma unroll
    for (int e = 0; e < 8; ++e) {
        const int o = wave * 16 + e + hoff;
        const float sc = g[o] * __frsqrt_rn(v[o] + EPSV);
        const float sh = be[o] - m[o] * sc;
        #pragma unroll
        for (int nt = 0; nt < 8; ++nt) {
            const int n = ntile * 128 + nt * 16 + (lane & 15);
            out[((size_t)b * CI + o) * NSP + n] =
                fmaxf(acc[nt][e] * sc + sh, 0.f);
        }
    }
}

// ===========================================================================
// Host-side launch
// ===========================================================================
extern "C" void kernel_launch(void* const* d_in, const int* in_sizes, int n_in,
                              void* d_out, int out_size, void* d_ws, size_t ws_size,
                              hipStream_t stream) {
    const float* q_in  = (const float*)d_in[0];
    const float* s_in  = (const float*)d_in[1];
    const float* w_v   = (const float*)d_in[2];
    const float* b_v   = (const float*)d_in[3];
    const float* w_k   = (const float*)d_in[4];
    const float* b_k   = (const float*)d_in[5];
    const float* w_q   = (const float*)d_in[6];
    const float* b_q   = (const float*)d_in[7];
    const float* w_ts  = (const float*)d_in[8];
    const float* b_ts  = (const float*)d_in[9];
    const float* bn_ts_g = (const float*)d_in[10];
    const float* bn_ts_b = (const float*)d_in[11];
    const float* bn_ts_m = (const float*)d_in[12];
    const float* bn_ts_v = (const float*)d_in[13];
    const float* w_tq  = (const float*)d_in[14];
    const float* b_tq  = (const float*)d_in[15];
    const float* bn_tq_g = (const float*)d_in[16];
    const float* bn_tq_b = (const float*)d_in[17];
    const float* bn_tq_m = (const float*)d_in[18];
    const float* bn_tq_v = (const float*)d_in[19];
    const float* cg1_w1 = (const float*)d_in[20];
    const float* cg1_b1 = (const float*)d_in[21];
    const float* cg1_w2 = (const float*)d_in[22];
    const float* cg1_b2 = (const float*)d_in[23];
    const float* cg2_w1 = (const float*)d_in[24];
    const float* cg2_b1 = (const float*)d_in[25];
    const float* cg2_w2 = (const float*)d_in[26];
    const float* cg2_b2 = (const float*)d_in[27];
    const float* w_cat  = (const float*)d_in[28];
    const float* bn_cat_g = (const float*)d_in[29];
    const float* bn_cat_b = (const float*)d_in[30];
    const float* bn_cat_m = (const float*)d_in[31];
    const float* bn_cat_v = (const float*)d_in[32];

    // workspace layout (~48 MiB total)
    char* ws = (char*)d_ws;
    unsigned short* kxT = (unsigned short*)(ws + ((size_t)0  << 20));
    unsigned short* qxT = (unsigned short*)(ws + ((size_t)4  << 20));
    unsigned short* vsT = (unsigned short*)(ws + ((size_t)8  << 20));
    unsigned short* vqT = (unsigned short*)(ws + ((size_t)12 << 20));
    float* s_t   = (float*)(ws + ((size_t)16 << 20));
    float* q_t   = (float*)(ws + ((size_t)24 << 20));
    float* p_s   = (float*)(ws + ((size_t)32 << 20));
    float* q_s   = (float*)(ws + ((size_t)40 << 20));
    float* gate1 = (float*)(ws + ((size_t)48 << 20));
    float* gate2 = gate1 + NB * CI;

    float* out = (float*)d_out;                       // [4,128,64,64]
    float* eq  = out + (size_t)NB * CI * NSP;         // E_q
    float* es  = eq  + (size_t)NB * CI * NSP;         // E_s

    k_conv1x1<<<dim3(32, 6, NB), 256, 0, stream>>>(
        q_in, s_in, w_v, b_v, w_k, b_k, w_q, b_q,
        w_ts, b_ts, bn_ts_g, bn_ts_b, bn_ts_m, bn_ts_v,
        w_tq, b_tq, bn_tq_g, bn_tq_b, bn_tq_m, bn_tq_v,
        vqT, vsT, kxT, qxT, s_t, q_t);

    k_gate<<<dim3(2, NB), 128, 0, stream>>>(
        vsT, vqT, cg1_w1, cg1_b1, cg1_w2, cg1_b2,
        cg2_w1, cg2_b1, cg2_w2, cg2_b2, gate1, gate2);

    k_attn<<<dim3(64, 2, NB), 128, 0, stream>>>(
        kxT, qxT, vsT, vqT, p_s, q_s);

    k_epilogue<<<dim3(8192, 2), 256, 0, stream>>>(
        p_s, q_s, s_t, q_t, gate1, gate2, eq, es);

    k_conv3x3<<<dim3(32, 1, NB), 256, 0, stream>>>(
        eq, es, w_cat, bn_cat_g, bn_cat_b, bn_cat_m, bn_cat_v, out);
}
